// ScaledDotProductAttention_12704513262289
// MI455X (gfx1250) — matrix operand order repaired and checked by reading, not compiled, not run on hardware
//
#include <hip/hip_runtime.h>
#include <hip/hip_bf16.h>
#include <stdint.h>

// ---------------------------------------------------------------------------
// SDPA for MI455X (gfx1250, wave32, WMMA).
// Shapes fixed by the reference: B=4, H=16, S=2048, DK=64, fp32 in/out.
// d_out = [context (B*H*S*DK floats)] ++ [attn_score (B*H*S*S floats)].
//
// HBM-write-bound on the 1.07 GB attn_score tensor -> write it exactly once
// (non-temporal, bypassing L2 residency), keep the full softmax row on-chip
// in a 16x2048 f32 LDS strip (128 KB of CDNA5's 320 KB WGP LDS).
// ---------------------------------------------------------------------------

typedef __attribute__((ext_vector_type(16))) _Float16 v16h;
typedef __attribute__((ext_vector_type(8)))  float    v8f;
typedef __attribute__((ext_vector_type(4)))  float    v4f;   // native vec for NT stores

#define S_LEN   2048
#define DK      64
#define NH      16
#define NB      4
#define QTILE   16
#define NEG_INF (-1000000000.0f)

__launch_bounds__(256, 1)
__global__ void sdpa_wmma_kernel(const float*   __restrict__ Q,
                                 const float*   __restrict__ K,
                                 const float*   __restrict__ V,
                                 const uint8_t* __restrict__ mask,
                                 float*         __restrict__ ctx_out,
                                 float*         __restrict__ score_out)
{
    // 128 KB score strip + small scratch: ~137 KB static LDS (CDNA5 WGP: 320 KB)
    __shared__ float sc[QTILE * S_LEN];      // 16 x 2048 f32 scores / probs
    __shared__ float qt[QTILE * DK];         // 16 x 64 Q tile
    __shared__ float red[QTILE * 16];        // per-row reduction scratch
    __shared__ float ctxred[4 * 16 * 16];    // PV cross-wave reduction

    const int tid  = threadIdx.x;
    const int lane = tid & 31;
    const int wave = tid >> 5;          // 0..7
    const int hi   = lane >> 4;         // half-wave: 0 = lanes 0-15, 1 = 16-31
    const int ln   = lane & 15;

    const int qtiles = S_LEN / QTILE;               // 128 (fastest dim -> L2 reuse of K/V)
    const int qtile  = blockIdx.x % qtiles;
    const int h      = (blockIdx.x / qtiles) % NH;
    const int b      =  blockIdx.x / (qtiles * NH);
    const int qbase  = qtile * QTILE;

    const size_t bh = (size_t)(b * NH + h);
    const float*   Qp = Q + (bh * S_LEN + (size_t)qbase) * DK;
    const float*   Kp = K + bh * S_LEN * DK;
    const float*   Vp = V + bh * S_LEN * DK;
    const uint8_t* Mp = mask + ((size_t)b * S_LEN + (size_t)qbase) * S_LEN;

    // ---- stage Q tile (16x64 f32 = 4 KB) via CDNA5 async LDS-direct load ----
    // 16 bytes per lane; tracked by ASYNCcnt (CDNA5 ISA).
    {
        const float*  src    = Qp + tid * 4;
        const uint32_t ldsoff = (uint32_t)(uintptr_t)(&qt[tid * 4]);
        asm volatile("global_load_async_to_lds_b128 %0, %1, off"
                     :: "v"(ldsoff), "v"(src) : "memory");
        asm volatile("s_wait_asynccnt 0x0" ::: "memory");
    }
    __syncthreads();

    // ---- build the two A fragments for Q (d = 0..31 and d = 32..63) ----
    // 16-bit A 16x32 layout: lane (l%16) = row M; half i -> K = (i%8) + 16*(i/8) + 8*hi
    v16h aq0, aq1;
    #pragma unroll
    for (int i = 0; i < 16; ++i) {
        const int kd = (i & 7) + ((i >> 3) << 4) + (hi << 3);     // 0..31
        aq0[i] = (_Float16)qt[ln * DK + kd];
        aq1[i] = (_Float16)qt[ln * DK + 32 + kd];
    }

    // ================= Phase 1: S = Q K^T (per-wave 16x16 tiles) =============
    // wave w owns key tiles [w*16, w*16+16)
    for (int kt = wave * 16; kt < wave * 16 + 16; ++kt) {
        const int kb = kt * 16;
        const float* kptr = Kp + (size_t)kb * DK;
        __builtin_prefetch(kptr + 16 * DK, 0, 1);   // global_prefetch_b8

        // 16-bit B 32x16 layout: lane (l%16) = col N; half i -> K-row = (i%8) + 16*(i/8) + 8*hi
        v16h b0, b1;
        #pragma unroll
        for (int i = 0; i < 16; ++i) {
            const int dr = (i & 7) + ((i >> 3) << 4) + (hi << 3);  // 0..31
            b0[i] = (_Float16)kptr[(size_t)ln * DK + dr];          // d in [0,32)
            b1[i] = (_Float16)kptr[(size_t)ln * DK + 32 + dr];     // d in [32,64)
        }
        v8f c = {};
        c = __builtin_amdgcn_wmma_f32_16x16x32_f16(false, aq0, false, b0,
                                                   (short)0, c, false, false);
        c = __builtin_amdgcn_wmma_f32_16x16x32_f16(false, aq1, false, b1,
                                                   (short)0, c, false, false);
        // f32 C/D layout: vgpr r -> row M = r + 8*hi, col N = lane%16
        #pragma unroll
        for (int r = 0; r < 8; ++r) {
            const int m = r + (hi << 3);
            sc[m * S_LEN + kb + ln] = c[r];
        }
    }
    __syncthreads();

    // ================= Phase 2: scale + mask + softmax (in LDS) ==============
    // 16 threads per row; thread `sub` handles 8 chunks of 16 elements at
    // base = c*256 + sub*16 so that the 16 threads of a row touch consecutive
    // 16B blocks (coalesced mask loads and score stores).
    const float scale = 0.125f;                 // 1/sqrt(64)
    const int row = tid >> 4;                   // 0..15
    const int sub = tid & 15;                   // 0..15
    const int q   = qbase + row;
    const uint8_t* mrow = Mp + (size_t)row * S_LEN;
    float* srow = sc + row * S_LEN;

    // issue all 8 coalesced 16-byte mask loads up front (one wait total)
    uint4 mb[8];
    #pragma unroll
    for (int c = 0; c < 8; ++c)
        mb[c] = *(const uint4*)(mrow + c * 256 + sub * 16);

    float lmax = -3.4e38f;
    #pragma unroll
    for (int c = 0; c < 8; ++c) {
        const int base = c * 256 + sub * 16;
        const uint32_t mw[4] = {mb[c].x, mb[c].y, mb[c].z, mb[c].w};
        #pragma unroll
        for (int j = 0; j < 4; ++j) {
            v4f s4 = *(v4f*)(srow + base + j * 4);
            s4.x = (mw[j] & 0x000000ffu) ? NEG_INF : s4.x * scale;
            s4.y = (mw[j] & 0x0000ff00u) ? NEG_INF : s4.y * scale;
            s4.z = (mw[j] & 0x00ff0000u) ? NEG_INF : s4.z * scale;
            s4.w = (mw[j] & 0xff000000u) ? NEG_INF : s4.w * scale;
            *(v4f*)(srow + base + j * 4) = s4;
            lmax = fmaxf(fmaxf(lmax, fmaxf(s4.x, s4.y)), fmaxf(s4.z, s4.w));
        }
    }
    red[row * 16 + sub] = lmax;
    __syncthreads();
    float rmax = red[row * 16];
    #pragma unroll
    for (int i = 1; i < 16; ++i) rmax = fmaxf(rmax, red[row * 16 + i]);
    __syncthreads();

    float lsum = 0.0f;
    #pragma unroll
    for (int c = 0; c < 8; ++c) {
        const int base = c * 256 + sub * 16;
        #pragma unroll
        for (int j = 0; j < 4; ++j) {
            v4f s4 = *(v4f*)(srow + base + j * 4);
            s4.x = __expf(s4.x - rmax);
            s4.y = __expf(s4.y - rmax);
            s4.z = __expf(s4.z - rmax);
            s4.w = __expf(s4.w - rmax);
            *(v4f*)(srow + base + j * 4) = s4;
            lsum += (s4.x + s4.y) + (s4.z + s4.w);
        }
    }
    red[row * 16 + sub] = lsum;
    __syncthreads();
    float rsum = 0.0f;
    #pragma unroll
    for (int i = 0; i < 16; ++i) rsum += red[row * 16 + i];
    const float rinv = 1.0f / rsum;

    // normalize in LDS; stream the probability row to HBM once, non-temporal
    // (write-once tensor: keep it out of the K/V/mask-hot L2)
    float* orow = score_out + ((size_t)(bh * S_LEN) + (size_t)q) * S_LEN;
    #pragma unroll
    for (int c = 0; c < 8; ++c) {
        const int base = c * 256 + sub * 16;
        #pragma unroll
        for (int j = 0; j < 4; ++j) {
            v4f p = *(v4f*)(srow + base + j * 4);
            p.x *= rinv; p.y *= rinv; p.z *= rinv; p.w *= rinv;
            *(v4f*)(srow + base + j * 4) = p;
            __builtin_nontemporal_store(p, (v4f*)(orow + base + j * 4));
        }
    }
    __syncthreads();

    // ================= Phase 3: context = P V (WMMA) =========================
    // wave -> (dblk = wave%4 covers d = dblk*16..+16, khalf = wave>>2 covers
    // k in [khalf*1024, khalf*1024+1024)); 32 WMMAs per wave.
    const int dblk  = wave & 3;
    const int khalf = wave >> 2;
    v8f acc = {};
    for (int kc = 0; kc < 32; ++kc) {
        const int kb = khalf * 1024 + kc * 32;
        const float* vptr = Vp + (size_t)kb * DK + dblk * 16;
        __builtin_prefetch(vptr + 32 * DK, 0, 1);

        v16h ap, bv;
        #pragma unroll
        for (int i = 0; i < 16; ++i) {
            const int kd = (i & 7) + ((i >> 3) << 4) + (hi << 3);   // A: K in chunk
            ap[i] = (_Float16)sc[ln * S_LEN + kb + kd];
            const int dr = (i & 7) + ((i >> 3) << 4) + (hi << 3);   // B: K-row
            bv[i] = (_Float16)vptr[(size_t)dr * DK + ln];
        }
        acc = __builtin_amdgcn_wmma_f32_16x16x32_f16(false, ap, false, bv,
                                                     (short)0, acc, false, false);
    }

    // reduce the two k-halves through LDS, then store context
    if (khalf == 1) {
        #pragma unroll
        for (int r = 0; r < 8; ++r) {
            const int m = r + (hi << 3);
            ctxred[(dblk * 16 + m) * 16 + ln] = acc[r];
        }
    }
    __syncthreads();
    if (khalf == 0) {
        float* crow = ctx_out + (bh * S_LEN + (size_t)qbase) * DK + dblk * 16;
        #pragma unroll
        for (int r = 0; r < 8; ++r) {
            const int m = r + (hi << 3);
            const float v2 = acc[r] + ctxred[(dblk * 16 + m) * 16 + ln];
            crow[(size_t)m * DK + ln] = v2;
        }
    }
}

extern "C" void kernel_launch(void* const* d_in, const int* in_sizes, int n_in,
                              void* d_out, int out_size, void* d_ws, size_t ws_size,
                              hipStream_t stream) {
    (void)in_sizes; (void)n_in; (void)out_size; (void)d_ws; (void)ws_size;

    const float*   Qp = (const float*)d_in[0];
    const float*   Kp = (const float*)d_in[1];
    const float*   Vp = (const float*)d_in[2];
    const uint8_t* Mp = (const uint8_t*)d_in[3];   // bool mask [B,1,S,S]
    // d_in[4] = dk scalar (always 64 for this reference) -> scale baked in.

    float* ctx   = (float*)d_out;                                   // [B,H,S,DK]
    float* score = (float*)d_out + (size_t)NB * NH * S_LEN * DK;    // [B,H,S,S]

    const int blocks = NB * NH * (S_LEN / QTILE);   // 8192 workgroups
    sdpa_wmma_kernel<<<blocks, 256, 0, stream>>>(Qp, Kp, Vp, Mp, ctx, score);
}
